// HGConv_48034914239360
// MI455X (gfx1250) — matrix operand order repaired and checked by reading, not compiled, not run on hardware
//
#include <hip/hip_runtime.h>

// ---------------------------------------------------------------------------
// HGConv forward for MI455X (gfx1250, wave32, WMMA).
// Dominant cost: inc_mat (268 MB f32) streamed once -> ~12us @ 23.3 TB/s.
// All GEMMs use v_wmma_f32_16x16x32_bf16 (bf16 A/B, f32 accum).
// ---------------------------------------------------------------------------

#define DIMV   300      // logical feature dim
#define DP     320      // padded feature dim (10 K-chunks of 32, 20 N-tiles of 16)
#define NTILE  20       // DP / 16
#define MN     16384    // nodes
#define EE     4096     // hyperedges

typedef __attribute__((ext_vector_type(16))) __bf16 v16bf;
typedef __attribute__((ext_vector_type(8)))  float  v8f;

union B16x16 {
  unsigned short u[16];
  unsigned       d[8];
  uint4          q[2];
  v16bf          v;
};

// exact round-to-nearest-even (cold pack paths)
__device__ __forceinline__ unsigned short f2bf(float f) {
  unsigned u = __float_as_uint(f);
  u += 0x7FFFu + ((u >> 16) & 1u);
  return (unsigned short)(u >> 16);
}

// fast path: bias + single v_perm_b32 packs two bf16 (3 VALU per pair)
__device__ __forceinline__ unsigned pkbf(float lo, float hi) {
  unsigned ul = __float_as_uint(lo) + 0x8000u;
  unsigned uh = __float_as_uint(hi) + 0x8000u;
  return __builtin_amdgcn_perm(uh, ul, 0x07060302u);  // {uh[31:16], ul[31:16]}
}

// A-operand K offset for vgpr v (0..7), lane-half hf (0/1):
// lanes 0-15: K = {0,2,4,...,14}; lanes 16-31: +16  (pairs K,K+1 per vgpr)
__device__ __forceinline__ int kbA(int v, int hf) {
  return 2 * v + (hf ? 16 : 0);
}

// ---------------------------------------------------------------------------
// Pack a row-major f32 matrix (srcRows x srcCols) into WMMA-B tiled layout:
// uint index = ((kc*NTILE + t)*32 + lane)*8 + v  holds bf16 pair
//   (src[kc*32 + 2v + (lane<16?0:16)][t*16 + lane%16],  next K row)
// ---------------------------------------------------------------------------
__global__ __launch_bounds__(256) void pack_B_f32(const float* __restrict__ src,
                                                  int srcRows, int srcCols,
                                                  unsigned* __restrict__ dst, int total) {
  int idx = blockIdx.x * 256 + threadIdx.x;
  if (idx >= total) return;
  int tile = idx >> 8;
  int kc = tile / NTILE, t = tile % NTILE;
  int w = idx & 255;
  int lane = w >> 3, v = w & 7;
  int hf = lane >> 4;
  int n = t * 16 + (lane & 15);
  int k = kc * 32 + 2 * v + (hf ? 16 : 0);
  float lo = (k     < srcRows && n < srcCols) ? src[(size_t)k * srcCols + n]       : 0.f;
  float hi = (k + 1 < srcRows && n < srcCols) ? src[(size_t)(k + 1) * srcCols + n] : 0.f;
  dst[idx] = ((unsigned)f2bf(hi) << 16) | (unsigned)f2bf(lo);
}

// Same, but source is already row-major bf16 with DP columns (no bounds needed).
__global__ __launch_bounds__(256) void pack_B_u16(const unsigned short* __restrict__ src,
                                                  unsigned* __restrict__ dst, int total) {
  int idx = blockIdx.x * 256 + threadIdx.x;
  if (idx >= total) return;
  int tile = idx >> 8;
  int kc = tile / NTILE, t = tile % NTILE;
  int w = idx & 255;
  int lane = w >> 3, v = w & 7;
  int hf = lane >> 4;
  int n = t * 16 + (lane & 15);
  int k = kc * 32 + 2 * v + (hf ? 16 : 0);
  unsigned lo = src[(size_t)k * DP + n];
  unsigned hi = src[(size_t)(k + 1) * DP + n];
  dst[idx] = (hi << 16) | lo;
}

__global__ __launch_bounds__(256) void zero_f32(float* __restrict__ p, int n) {
  int i = blockIdx.x * 256 + threadIdx.x;
  if (i < n) p[i] = 0.f;
}

// ---------------------------------------------------------------------------
// gemm1: Ab (MN x DP, bf16 row-major) = node_feats (MN x 300 f32) @ W_att (packed B)
// One 16x16 tile per wave; 10 K-chunks of 32.
// ---------------------------------------------------------------------------
__global__ __launch_bounds__(256) void gemm_nodeW(const float* __restrict__ node,
                                                  const unsigned* __restrict__ Wpack,
                                                  unsigned short* __restrict__ Ab) {
  int wave = blockIdx.x * 8 + (threadIdx.x >> 5);
  int lane = threadIdx.x & 31;
  int rowTile = wave / NTILE, t = wave % NTILE;
  int m0 = rowTile * 16, n0 = t * 16;
  int hf = lane >> 4, r = lane & 15;
  const float* rowp = node + (size_t)(m0 + r) * DIMV;
  v8f acc = {0.f, 0.f, 0.f, 0.f, 0.f, 0.f, 0.f, 0.f};
  for (int kc = 0; kc < DP / 32; ++kc) {
    int k0 = kc * 32;
    B16x16 A;
#pragma unroll
    for (int v = 0; v < 8; ++v) {
      int k = k0 + kbA(v, hf);
      float f0 = (k     < DIMV) ? rowp[k]     : 0.f;
      float f1 = (k + 1 < DIMV) ? rowp[k + 1] : 0.f;
      A.d[v] = pkbf(f0, f1);
    }
    B16x16 B;
    const uint4* bp = (const uint4*)(Wpack + ((size_t)(kc * NTILE + t) * 32 + lane) * 8);
    B.q[0] = bp[0];
    B.q[1] = bp[1];
    acc = __builtin_amdgcn_wmma_f32_16x16x32_bf16(false, A.v, false, B.v, (short)0, acc,
                                                  false, false);
  }
#pragma unroll
  for (int v = 0; v < 8; ++v)
    Ab[(size_t)(m0 + v + hf * 8) * DP + n0 + r] = f2bf(acc[v]);
}

// ---------------------------------------------------------------------------
// gemm2 (the 268MB pass): ES = inc.T @ X ; EA = inc.T @ (X@W_att), fused.
// Block = 8 waves sharing one 16-edge strip + one K-segment (inc tile loads
// merge in WGP$ -> inc streamed from HBM once). Wave w owns 5 of 40 col
// tiles (w<4 -> Xp/ES, w>=4 -> Ap/EA). Split-K x8 via f32 atomics.
// ---------------------------------------------------------------------------
__global__ __launch_bounds__(256) void gemm_inc(const float* __restrict__ inc,
                                                const unsigned* __restrict__ Xp,
                                                const unsigned* __restrict__ Ap,
                                                float* __restrict__ ES,
                                                float* __restrict__ EA) {
  int w = threadIdx.x >> 5, lane = threadIdx.x & 31;
  int eTile = blockIdx.x & 255;   // 256 edge strips
  int kseg  = blockIdx.x >> 8;    // 8 K segments
  int e0 = eTile * 16;
  int hf = lane >> 4, r = lane & 15;
  int e = e0 + r;
  const unsigned* src = (w < 4) ? Xp : Ap;
  float*          dst = (w < 4) ? ES : EA;
  int tbase = (w * 5) % NTILE;

  v8f acc[5];
#pragma unroll
  for (int j = 0; j < 5; ++j) acc[j] = (v8f){0.f, 0.f, 0.f, 0.f, 0.f, 0.f, 0.f, 0.f};

  int kcBeg = kseg * 64, kcEnd = kcBeg + 64;   // 512 chunks total / 8 segments
  for (int kc = kcBeg; kc < kcEnd; ++kc) {
    int m0 = kc * 32;
    if (kc + 1 < 512)  // pull next inc chunk (emits global_prefetch_b8)
      __builtin_prefetch(inc + (size_t)(m0 + 32 + kbA(0, hf)) * EE + e, 0, 1);
    B16x16 A;   // A[i=edge, k=node] = inc[m0+k][e0+i], fast bf16 pack via v_perm
#pragma unroll
    for (int v = 0; v < 8; ++v) {
      const float* p = inc + (size_t)(m0 + kbA(v, hf)) * EE + e;
      A.d[v] = pkbf(p[0], p[EE]);
    }
#pragma unroll
    for (int j = 0; j < 5; ++j) {
      int t = tbase + j;
      B16x16 B;
      const uint4* bp = (const uint4*)(src + ((size_t)(kc * NTILE + t) * 32 + lane) * 8);
      B.q[0] = bp[0];
      B.q[1] = bp[1];
      acc[j] = __builtin_amdgcn_wmma_f32_16x16x32_bf16(false, A.v, false, B.v, (short)0,
                                                       acc[j], false, false);
    }
  }
#pragma unroll
  for (int j = 0; j < 5; ++j) {
    int n0 = (tbase + j) * 16;
#pragma unroll
    for (int v = 0; v < 8; ++v)
      atomicAdd(&dst[(size_t)(e0 + v + hf * 8) * DP + n0 + r], acc[j][v]);
  }
}

// ---------------------------------------------------------------------------
// Per-edge softmax over D of EA, multiply into ES, emit Gb (bf16 row-major).
// ---------------------------------------------------------------------------
__global__ __launch_bounds__(DP) void edge_softmax_mul(const float* __restrict__ EA,
                                                       const float* __restrict__ ES,
                                                       unsigned short* __restrict__ Gb) {
  __shared__ float red[DP];
  __shared__ float smx, ssum;
  int e = blockIdx.x, d = threadIdx.x;
  float val = (d < DIMV) ? EA[(size_t)e * DP + d] : -3.0e38f;
  red[d] = val;
  __syncthreads();
  if (d < 32) {
    float m = red[d];
    for (int i = d + 32; i < DP; i += 32) m = fmaxf(m, red[i]);
    for (int off = 16; off; off >>= 1) m = fmaxf(m, __shfl_xor(m, off, 32));
    if (d == 0) smx = m;
  }
  __syncthreads();
  float ex = (d < DIMV) ? __expf(val - smx) : 0.f;
  red[d] = ex;
  __syncthreads();
  if (d < 32) {
    float s = red[d];
    for (int i = d + 32; i < DP; i += 32) s += red[i];
    for (int off = 16; off; off >>= 1) s += __shfl_xor(s, off, 32);
    if (d == 0) ssum = s;
  }
  __syncthreads();
  float g = (d < DIMV) ? ES[(size_t)e * DP + d] * (ex / ssum) : 0.f;
  Gb[(size_t)e * DP + d] = f2bf(g);
}

// ---------------------------------------------------------------------------
// gemm3: EF = alpha*edge_feats + (1-alpha)*(Gb @ W_proj)   (EE x DP, f32)
// ---------------------------------------------------------------------------
__global__ __launch_bounds__(256) void gemm_proj(const unsigned short* __restrict__ Gb,
                                                 const unsigned* __restrict__ Wpack,
                                                 const float* __restrict__ edge_feats,
                                                 const float* __restrict__ alpha_p,
                                                 float* __restrict__ EF) {
  int wave = blockIdx.x * 8 + (threadIdx.x >> 5);
  int lane = threadIdx.x & 31;
  int rowTile = wave / NTILE, t = wave % NTILE;
  int m0 = rowTile * 16, n0 = t * 16;
  int hf = lane >> 4, r = lane & 15;
  const unsigned short* rowp = Gb + (size_t)(m0 + r) * DP;
  v8f acc = {0.f, 0.f, 0.f, 0.f, 0.f, 0.f, 0.f, 0.f};
  for (int kc = 0; kc < DP / 32; ++kc) {
    int k0 = kc * 32;
    B16x16 A;
#pragma unroll
    for (int v = 0; v < 8; ++v) {
      int k = k0 + kbA(v, hf);          // even -> dword-aligned bf16 pair
      A.d[v] = *(const unsigned*)(rowp + k);
    }
    B16x16 B;
    const uint4* bp = (const uint4*)(Wpack + ((size_t)(kc * NTILE + t) * 32 + lane) * 8);
    B.q[0] = bp[0];
    B.q[1] = bp[1];
    acc = __builtin_amdgcn_wmma_f32_16x16x32_bf16(false, A.v, false, B.v, (short)0, acc,
                                                  false, false);
  }
  float al = *alpha_p;
#pragma unroll
  for (int v = 0; v < 8; ++v) {
    int e = m0 + v + hf * 8;
    int col = n0 + r;
    float res = (col < DIMV) ? al * edge_feats[(size_t)e * DIMV + col] : 0.f;
    EF[(size_t)e * DP + col] = res + (1.f - al) * acc[v];
  }
}

// scores[e] = EF[e,:300] . ec_W_att  — one wave per edge
__global__ __launch_bounds__(256) void edge_scores(const float* __restrict__ EF,
                                                   const float* __restrict__ watt,
                                                   float* __restrict__ scores) {
  int e = blockIdx.x * 8 + (threadIdx.x >> 5);
  int lane = threadIdx.x & 31;
  float s = 0.f;
  for (int d = lane; d < DIMV; d += 32) s += EF[(size_t)e * DP + d] * watt[d];
  for (int off = 16; off; off >>= 1) s += __shfl_xor(s, off, 32);
  if (lane == 0) scores[e] = s;
}

// softmax stats over 4096 edge scores
__global__ __launch_bounds__(1024) void score_stats(const float* __restrict__ scores,
                                                    float* __restrict__ stats) {
  __shared__ float red[1024];
  int t = threadIdx.x;
  float m = -3.0e38f;
  for (int i = t; i < EE; i += 1024) m = fmaxf(m, scores[i]);
  red[t] = m;
  __syncthreads();
  for (int s = 512; s; s >>= 1) { if (t < s) red[t] = fmaxf(red[t], red[t + s]); __syncthreads(); }
  float mx = red[0];
  __syncthreads();
  float sm = 0.f;
  for (int i = t; i < EE; i += 1024) sm += __expf(scores[i] - mx);
  red[t] = sm;
  __syncthreads();
  for (int s = 512; s; s >>= 1) { if (t < s) red[t] += red[t + s]; __syncthreads(); }
  if (t == 0) { stats[0] = mx; stats[1] = red[0]; }
}

// pooled[d] += sum over a 64-edge slab of EF[e,d]*softmax_w[e]
__global__ __launch_bounds__(DP) void pooled_partial(const float* __restrict__ EF,
                                                     const float* __restrict__ scores,
                                                     const float* __restrict__ stats,
                                                     float* __restrict__ pooled) {
  int d = threadIdx.x;
  int e0 = blockIdx.x * 64;
  float mx = stats[0], den = stats[1];
  float s = 0.f;
  for (int e = e0; e < e0 + 64; ++e)
    s += EF[(size_t)e * DP + d] * (__expf(scores[e] - mx) / den);
  atomicAdd(&pooled[d], s);
}

// out = pooled @ ec_W_proj + ec_b_proj ; logits = out @ fc_W + fc_b
__global__ __launch_bounds__(DP) void final_head(const float* __restrict__ pooled,
                                                 const float* __restrict__ Wp,
                                                 const float* __restrict__ bp,
                                                 const float* __restrict__ fcW,
                                                 const float* __restrict__ fcb,
                                                 float* __restrict__ logits) {
  __shared__ float o[DIMV];
  int j = threadIdx.x;
  if (j < DIMV) {
    float s = bp[j];
    for (int i = 0; i < DIMV; ++i) s += pooled[i] * Wp[(size_t)i * DIMV + j];
    o[j] = s;
  }
  __syncthreads();
  if (j < 3) {
    float s = fcb[j];
    for (int i = 0; i < DIMV; ++i) s += o[i] * fcW[i * 3 + j];
    logits[j] = s;
  }
}

// ---------------------------------------------------------------------------
extern "C" void kernel_launch(void* const* d_in, const int* in_sizes, int n_in,
                              void* d_out, int out_size, void* d_ws, size_t ws_size,
                              hipStream_t stream) {
  const float* node  = (const float*)d_in[0];
  const float* edgef = (const float*)d_in[1];
  const float* inc   = (const float*)d_in[2];
  const float* Watt  = (const float*)d_in[3];
  const float* Wproj = (const float*)d_in[4];
  const float* alpha = (const float*)d_in[5];
  const float* ecWa  = (const float*)d_in[6];
  const float* ecWp  = (const float*)d_in[7];
  const float* ecb   = (const float*)d_in[8];
  const float* fcW   = (const float*)d_in[9];
  const float* fcb   = (const float*)d_in[10];

  // workspace layout (all chunks multiples of 256B)
  const size_t XP_B  = (size_t)512 * NTILE * 256 * 4;  // 10,485,760
  const size_t AP_B  = XP_B;
  const size_t WA_B  = (size_t)10 * NTILE * 256 * 4;   // 204,800
  const size_t WP_B  = WA_B;
  const size_t AB_B  = (size_t)MN * DP * 2;            // 10,485,760
  const size_t GB_B  = (size_t)EE * DP * 2;            // 2,621,440
  const size_t EF_B  = (size_t)EE * DP * 4;            // 5,242,880
  const size_t SC_B  = (size_t)EE * 4;                 // 16,384
  const size_t ST_B  = 256;
  const size_t ES_B  = (size_t)EE * DP * 4;
  const size_t EA_B  = ES_B;
  const size_t PL_B  = (size_t)DP * 4;                 // 1,280

  char* ws = (char*)d_ws;
  unsigned*       Xp     = (unsigned*)(ws);
  unsigned*       Ap     = (unsigned*)(ws + XP_B);
  unsigned*       WaP    = (unsigned*)(ws + XP_B + AP_B);
  unsigned*       WpP    = (unsigned*)(ws + XP_B + AP_B + WA_B);
  unsigned short* Ab     = (unsigned short*)(ws + XP_B + AP_B + WA_B + WP_B);
  unsigned short* Gb     = (unsigned short*)(ws + XP_B + AP_B + WA_B + WP_B + AB_B);
  float*          EF     = (float*)(ws + XP_B + AP_B + WA_B + WP_B + AB_B + GB_B);
  float*          scores = (float*)(ws + XP_B + AP_B + WA_B + WP_B + AB_B + GB_B + EF_B);
  float*          stats  = (float*)(ws + XP_B + AP_B + WA_B + WP_B + AB_B + GB_B + EF_B + SC_B);
  float*          ES     = (float*)(ws + XP_B + AP_B + WA_B + WP_B + AB_B + GB_B + EF_B + SC_B + ST_B);
  float*          EA     = (float*)((char*)ES + ES_B);
  float*          pooled = (float*)((char*)ES + ES_B + EA_B);
  (void)n_in; (void)in_sizes; (void)out_size; (void)ws_size;

  // 0) zero the atomic-accumulated buffers (ES, EA, pooled are contiguous)
  {
    int nz = (int)((ES_B + EA_B + PL_B) / 4);
    zero_f32<<<(nz + 255) / 256, 256, 0, stream>>>(ES, nz);
  }
  // 1) pack weights + node feats into WMMA-B layout
  pack_B_f32<<<200, 256, 0, stream>>>(Watt, DIMV, DIMV, WaP, 10 * NTILE * 256);
  pack_B_f32<<<200, 256, 0, stream>>>(Wproj, DIMV, DIMV, WpP, 10 * NTILE * 256);
  pack_B_f32<<<10240, 256, 0, stream>>>(node, MN, DIMV, Xp, 512 * NTILE * 256);
  // 2) Ab = node @ W_att (bf16 row-major), then repack to B layout
  gemm_nodeW<<<2560, 256, 0, stream>>>(node, WaP, Ab);
  pack_B_u16<<<10240, 256, 0, stream>>>(Ab, Ap, 512 * NTILE * 256);
  // 3) ES = inc.T @ X, EA = inc.T @ (X@W_att)  (fused, split-K atomics)
  gemm_inc<<<2048, 256, 0, stream>>>(inc, Xp, Ap, ES, EA);
  // 4) attention softmax over D, G = ES * attn (bf16)
  edge_softmax_mul<<<EE, DP, 0, stream>>>(EA, ES, Gb);
  // 5) EF = alpha*edge_feats + (1-alpha)*(G @ W_proj)
  gemm_proj<<<640, 256, 0, stream>>>(Gb, WpP, edgef, alpha, EF);
  // 6) edge aggregation: softmax over edges, weighted pool
  edge_scores<<<512, 256, 0, stream>>>(EF, ecWa, scores);
  score_stats<<<1, 1024, 0, stream>>>(scores, stats);
  pooled_partial<<<64, DP, 0, stream>>>(EF, scores, stats, pooled);
  // 7) head
  final_head<<<1, DP, 0, stream>>>(pooled, ecWp, ecb, fcW, fcb, (float*)d_out);
}
